// ParamLstm_9165460210023
// MI455X (gfx1250) — matrix operand order repaired and checked by reading, not compiled, not run on hardware
//
#include <hip/hip_runtime.h>

typedef __bf16 bf16_t;
typedef unsigned int u32;
typedef __attribute__((ext_vector_type(16))) __bf16 v16bf;
typedef __attribute__((ext_vector_type(8)))  float  v8f;

#define NI      16384   // N_INSTR
#define TT      16      // T_TOK
#define NB      512     // N_BLOCKS
#define MAXB    64      // MAX_BLK
#define H       128
#define G4H     512     // 4*H
#define P_INSTR 32
#define P_GLOB  8
#define N_TOKENS 1024
#define D_WORD  296
#define D_PAD   320     // D_WORD padded to multiple of 32 for WMMA K

// ---------------------------------------------------------------- WMMA helpers

__device__ __forceinline__ v8f wmma_bf16(v16bf a, v16bf b, v8f c) {
  // D = A(16x32) * B(32x16) + C, fp32 accumulate
  return __builtin_amdgcn_wmma_f32_16x16x32_bf16(
      false, a, false, b, (short)0, c, false, false);
}

// A fragment: 16xK tile, rows 0..15 of row-major [16 x stride] bf16 matrix,
// K window = kc*32 .. kc*32+31.  gfx1250 16-bit A layout:
//  lane<16 (M=lane):  dw0..3 = K(0,1)(2,3)(4,5)(6,7);  dw4..7 = K(16..23)
//  lane>=16(M=l-16):  dw0..3 = K(8..15);               dw4..7 = K(24..31)
// Per lane: two contiguous 16B runs -> two b128 loads.
__device__ __forceinline__ v16bf load_frag_a(const bf16_t* base, int stride,
                                             int kc, int lane) {
  const int M   = lane & 15;
  const int kb2 = (lane < 16) ? 0 : 4;   // dword offset
  const uint4* p = (const uint4*)((const u32*)(base + M * stride + kc * 32) + kb2);
  union { v16bf v; uint4 q[2]; } f;
  f.q[0] = p[0];      // dwords kb2 .. kb2+3
  f.q[1] = p[2];      // dwords kb2+8 .. kb2+11
  return f.v;
}

// Same, but this lane's matrix row is given directly (per-lane row pointer).
__device__ __forceinline__ v16bf load_frag_a_row(const bf16_t* row, int kc, int lane) {
  const int kb2 = (lane < 16) ? 0 : 4;
  const uint4* p = (const uint4*)((const u32*)(row + kc * 32) + kb2);
  union { v16bf v; uint4 q[2]; } f;
  f.q[0] = p[0];
  f.q[1] = p[2];
  return f.v;
}

// B fragment: B = W^T slice (32x16). W is row-major [512 x stride] bf16,
// row = output column, col = k.  gfx1250 16-bit B layout:
//  lanes 0-15: K(0..7), K(16..23) (N=lane); lanes 16-31: K(8..15), K(24..31) (N=lane-16)
// Per lane: two contiguous 16B runs -> two b128 loads.
__device__ __forceinline__ v16bf load_frag_b(const bf16_t* W, int stride,
                                             int colbase, int kc, int lane) {
  const int N  = lane & 15;
  const int kr = (lane < 16) ? 0 : 8;
  const uint4* p = (const uint4*)(W + (size_t)(colbase + N) * stride + kc * 32 + kr);
  union { v16bf v; uint4 q[2]; } f;
  f.q[0] = p[0];
  f.q[1] = p[2];
  return f.v;
}

__device__ __forceinline__ float sigm(float x) {
  return 1.0f / (1.0f + __expf(-x));
}
__device__ __forceinline__ float tanh_f(float x) {
  return 2.0f * sigm(2.0f * x) - 1.0f;
}

// ---------------------------------------------------------------- small kernels

__global__ void k_cvt(const float* __restrict__ src, bf16_t* __restrict__ dst, int n) {
  int i = blockIdx.x * blockDim.x + threadIdx.x;
  if (i < n) dst[i] = (bf16_t)src[i];
}

// pad 512x296 -> 512x320 with zeros
__global__ void k_cvt_pad(const float* __restrict__ src, bf16_t* __restrict__ dst,
                          int rows, int scol, int dcol) {
  int i = blockIdx.x * blockDim.x + threadIdx.x;
  int r = i / dcol, c = i % dcol;
  if (r < rows) dst[i] = (c < scol) ? (bf16_t)src[(size_t)r * scol + c] : (bf16_t)0.0f;
}

// inclusive scan of block_lens (512 entries, one block)
__global__ void k_scan(const int* __restrict__ block_lens,
                       int* __restrict__ starts, int* __restrict__ ends) {
  __shared__ int s[NB];
  int tid = threadIdx.x;
  s[tid] = block_lens[tid];
  __syncthreads();
  for (int d = 1; d < NB; d <<= 1) {
    int v = (tid >= d) ? s[tid - d] : 0;
    __syncthreads();
    s[tid] += v;
    __syncthreads();
  }
  ends[tid]   = s[tid];
  starts[tid] = s[tid] - block_lens[tid];
}

// full[i][256..319] = {instr_params, global_params[bid], zeros}  (bf16)
__global__ void k_fill(const float* __restrict__ instr_params,
                       const float* __restrict__ global_params,
                       const int* __restrict__ ends,
                       bf16_t* __restrict__ full) {
  int i = blockIdx.x;
  int j = threadIdx.x;                 // 0..63
  // bid = searchsorted(ends, i, 'right')
  int lo = 0, hi = NB;
  while (lo < hi) { int mid = (lo + hi) >> 1; if (ends[mid] <= i) lo = mid + 1; else hi = mid; }
  int bid = lo;
  float v = 0.0f;
  if (j < P_INSTR)               v = instr_params[(size_t)i * P_INSTR + j];
  else if (j < P_INSTR + P_GLOB) v = global_params[(size_t)bid * P_GLOB + (j - P_INSTR)];
  full[(size_t)i * D_PAD + 2 * H + j] = (bf16_t)v;
}

// ---------------------------------------------------------------- char BiLSTM
// grid (NI/128, 2=dir), block 256 (8 waves -> 2 waves/SIMD for WMMA/VALU overlap).
// One wave = 16 instructions. Wih+Whh (bf16) LDS-resident, shared by 8 waves.
// x_t A-fragments are loaded straight from the bf16 embedding table (2 b128
// global loads per fragment) -- no x staging, no extra barrier.

__global__ void k_char_lstm(const int* __restrict__ tokens,
                            const int* __restrict__ token_lens,
                            const bf16_t* __restrict__ emb_bf,    // [1024][128]
                            const bf16_t* __restrict__ wih_all,   // [2][512*128]
                            const bf16_t* __restrict__ whh_all,   // [2][512*128]
                            const float* __restrict__ b_f,
                            const float* __restrict__ b_b,
                            bf16_t* __restrict__ full) {
  extern __shared__ char smem_raw[];
  bf16_t* s_wih  = (bf16_t*)smem_raw;              // 512*128
  bf16_t* s_whh  = s_wih + G4H * H;                // 512*128
  float*  s_bias = (float*)(s_whh + G4H * H);      // 512
  bf16_t* s_h    = (bf16_t*)(s_bias + G4H);        // 8 * 16*128
  int*    s_len  = (int*)(s_h + 8 * 16 * H);       // 8 * 16

  const int dir  = blockIdx.y;
  const int wave = threadIdx.x >> 5;
  const int lane = threadIdx.x & 31;
  const int tid  = threadIdx.x;

  const bf16_t* wih  = wih_all + (size_t)dir * (G4H * H);
  const bf16_t* whh  = whh_all + (size_t)dir * (G4H * H);
  const float*  bias = dir ? b_b : b_f;

  __builtin_prefetch(wih, 0, 1);
  __builtin_prefetch(whh, 0, 1);

  // cooperative weight staging (uint4 = 8 bf16 per transaction)
  {
    const uint4* src = (const uint4*)wih;
    uint4* dst = (uint4*)s_wih;
    for (int i = tid; i < G4H * H / 8; i += blockDim.x) dst[i] = src[i];
    src = (const uint4*)whh; dst = (uint4*)s_whh;
    for (int i = tid; i < G4H * H / 8; i += blockDim.x) dst[i] = src[i];
    for (int i = tid; i < G4H; i += blockDim.x) s_bias[i] = bias[i];
  }

  const int ibase = (blockIdx.x * 8 + wave) * 16;
  bf16_t* hw   = s_h + wave * 16 * H;
  int*    lenw = s_len + wave * 16;
  if (lane < 16) lenw[lane] = token_lens[ibase + lane];
  { uint4 z = {0u, 0u, 0u, 0u};
    uint4* hz = (uint4*)hw;
    for (int i = lane; i < 16 * H / 8; i += 32) hz[i] = z; }
  __syncthreads();

  const int Ml  = lane & 15;             // this lane's A-fragment row
  const int off = (lane < 16) ? 0 : 8;   // C-fragment row offset
  const int lenM = lenw[Ml];
  int lenv[8];
#pragma unroll
  for (int r = 0; r < 8; ++r) lenv[r] = lenw[r + off];

  v8f cc[8];
#pragma unroll
  for (int n = 0; n < 8; ++n)
#pragma unroll
    for (int e = 0; e < 8; ++e) cc[n][e] = 0.0f;

  for (int t = 0; t < TT; ++t) {
    // per-lane token row for time t (reversed + clamped for backward dir)
    int st = (dir == 0) ? t : (lenM - 1 - t);
    st = st < 0 ? 0 : (st > TT - 1 ? TT - 1 : st);
    int tok = tokens[(size_t)(ibase + Ml) * TT + st];
    const bf16_t* xrow = emb_bf + (size_t)tok * H;

    v16bf ax[4], ah[4];
#pragma unroll
    for (int kc = 0; kc < 4; ++kc) {
      ax[kc] = load_frag_a_row(xrow, kc, lane);   // global b128 x2
      ah[kc] = load_frag_a(hw, H, kc, lane);      // ds b128 x2
    }

#pragma unroll 1
    for (int n = 0; n < 8; ++n) {            // 8 tiles of 16 H-columns
      v8f acc[4];                            // i, f, g, o tiles
#pragma unroll
      for (int q = 0; q < 4; ++q) {
        const int cb = q * H + n * 16;
        float bv = s_bias[cb + (lane & 15)];
#pragma unroll
        for (int e = 0; e < 8; ++e) acc[q][e] = bv;
#pragma unroll
        for (int kc = 0; kc < 4; ++kc) {
          acc[q] = wmma_bf16(ax[kc], load_frag_b(s_wih, H, cb, kc, lane), acc[q]);
          acc[q] = wmma_bf16(ah[kc], load_frag_b(s_whh, H, cb, kc, lane), acc[q]);
        }
      }
      // elementwise LSTM cell in C-fragment layout
#pragma unroll
      for (int r = 0; r < 8; ++r) {
        if (t < lenv[r]) {
          float iv = sigm(acc[0][r]);
          float fv = sigm(acc[1][r]);
          float gv = tanh_f(acc[2][r]);
          float ov = sigm(acc[3][r]);
          float cn = fv * cc[n][r] + iv * gv;
          cc[n][r] = cn;
          float hn = ov * tanh_f(cn);
          hw[(r + off) * H + n * 16 + (lane & 15)] = (bf16_t)hn;
        }
      }
    }
    __syncthreads();
  }

  // final h -> full[:, dir*H .. dir*H+128) (bf16, row stride D_PAD), 16B chunks
  const uint4* hsrc = (const uint4*)hw;
  for (int i = lane; i < 16 * H / 8; i += 32) {
    int e = 8 * i;
    int m = e / H, col = e % H;
    *(uint4*)(full + (size_t)(ibase + m) * D_PAD + dir * H + col) = hsrc[i];
  }
}

// ---------------------------------------------------------------- word input GEMM
// xW[dir] = full(16384 x 320 bf16) @ Wih[dir]^T + b[dir]   -> fp32 (16384 x 512)
// grid (NI/16, 8, 2), block 128: 4 waves share one LDS-staged 16x320 A tile,
// each wave produces one 16x16 output tile. K = 320 (10 WMMAs).

__global__ void k_word_gemm(const bf16_t* __restrict__ full,
                            const bf16_t* __restrict__ wih_all,  // [2][512*320]
                            const float* __restrict__ b_f,
                            const float* __restrict__ b_b,
                            float* __restrict__ xW_all) {
  __shared__ bf16_t sA[16 * D_PAD];          // 10 KB
  const int dir  = blockIdx.z;
  const int wave = threadIdx.x >> 5;
  const int lane = threadIdx.x & 31;
  const int mt   = blockIdx.x;
  const int nt   = blockIdx.y * 4 + wave;
  const bf16_t* W    = wih_all + (size_t)dir * (G4H * D_PAD);
  const float*  bias = dir ? b_b : b_f;
  float* xW = xW_all + (size_t)dir * ((size_t)NI * G4H);
  const int cb = nt * 16;

  // stage shared A tile
  {
    const uint4* src = (const uint4*)(full + (size_t)mt * 16 * D_PAD);
    uint4* dst = (uint4*)sA;
    for (int i = threadIdx.x; i < 16 * D_PAD / 8; i += blockDim.x) dst[i] = src[i];
  }
  __syncthreads();

  v8f acc;
  float bv = bias[cb + (lane & 15)];
#pragma unroll
  for (int e = 0; e < 8; ++e) acc[e] = bv;

#pragma unroll 1
  for (int kc = 0; kc < D_PAD / 32; ++kc) {
    v16bf a = load_frag_a(sA, D_PAD, kc, lane);
    v16bf b = load_frag_b(W, D_PAD, cb, kc, lane);
    acc = wmma_bf16(a, b, acc);
  }

  const int off = (lane < 16) ? 0 : 8;
#pragma unroll
  for (int r = 0; r < 8; ++r)
    xW[(size_t)(mt * 16 + r + off) * G4H + cb + (lane & 15)] = acc[r];
}

// ---------------------------------------------------------------- word BiLSTM
// grid (32, 2=dir), block 32 (1 wave per WGP -> full matrix pipe per wave;
// parallelism is capped at 64 waves so spread them across WGPs).
// Gates = xW[row(t)] (precomputed, incl. bias) + h @ Whh^T (LDS-resident).

__global__ void k_word_lstm(const bf16_t* __restrict__ whh_all,  // [2][512*128]
                            const float* __restrict__ xW_all,    // [2][NI*512]
                            const int* __restrict__ starts,
                            const int* __restrict__ block_lens,
                            float* __restrict__ blockfeat) {     // [512][256]
  extern __shared__ char smem_raw[];
  bf16_t* s_whh = (bf16_t*)smem_raw;              // 512*128
  bf16_t* hw    = s_whh + G4H * H;                // 16*128
  int*    stw   = (int*)(hw + 16 * H);            // 16
  int*    lnw   = stw + 16;                       // 16

  const int dir  = blockIdx.y;
  const int lane = threadIdx.x & 31;
  const bf16_t* whh = whh_all + (size_t)dir * (G4H * H);
  const float*  xW  = xW_all + (size_t)dir * ((size_t)NI * G4H);

  {
    const uint4* src = (const uint4*)whh;
    uint4* dst = (uint4*)s_whh;
    for (int i = lane; i < G4H * H / 8; i += 32) dst[i] = src[i];
  }

  const int bbase = blockIdx.x * 16;
  if (lane < 16) { stw[lane] = starts[bbase + lane]; lnw[lane] = block_lens[bbase + lane]; }
  { uint4 z = {0u, 0u, 0u, 0u};
    uint4* hz = (uint4*)hw;
    for (int i = lane; i < 16 * H / 8; i += 32) hz[i] = z; }
  __syncthreads();

  const int off = (lane < 16) ? 0 : 8;
  int st8[8], ln8[8];
#pragma unroll
  for (int r = 0; r < 8; ++r) { st8[r] = stw[r + off]; ln8[r] = lnw[r + off]; }

  v8f cc[8];
#pragma unroll
  for (int n = 0; n < 8; ++n)
#pragma unroll
    for (int e = 0; e < 8; ++e) cc[n][e] = 0.0f;

  for (int t = 0; t < MAXB; ++t) {
    int idx8[8];
#pragma unroll
    for (int r = 0; r < 8; ++r) {
      int len = ln8[r];
      int s = (dir == 0) ? t : (len - 1 - t);
      s = s < 0 ? 0 : (s >= len ? len - 1 : s);
      idx8[r] = st8[r] + s;
    }
    // prefetch next timestep's gate rows (global_prefetch_b8)
    {
      int tn = t + 1;
      if (tn < MAXB) {
#pragma unroll
        for (int r = 0; r < 8; ++r) {
          int len = ln8[r];
          int s = (dir == 0) ? tn : (len - 1 - tn);
          s = s < 0 ? 0 : (s >= len ? len - 1 : s);
          __builtin_prefetch(xW + (size_t)(st8[r] + s) * G4H + lane * 16, 0, 1);
        }
      }
    }

    v16bf ah[4];
#pragma unroll
    for (int kc = 0; kc < 4; ++kc) ah[kc] = load_frag_a(hw, H, kc, lane);

#pragma unroll 1
    for (int n = 0; n < 8; ++n) {
      v8f acc[4];
#pragma unroll
      for (int q = 0; q < 4; ++q) {
        const int cb = q * H + n * 16;
        const int col = cb + (lane & 15);
#pragma unroll
        for (int r = 0; r < 8; ++r)
          acc[q][r] = xW[(size_t)idx8[r] * G4H + col];
#pragma unroll
        for (int kc = 0; kc < 4; ++kc)
          acc[q] = wmma_bf16(ah[kc], load_frag_b(s_whh, H, cb, kc, lane), acc[q]);
      }
#pragma unroll
      for (int r = 0; r < 8; ++r) {
        if (t < ln8[r]) {
          float iv = sigm(acc[0][r]);
          float fv = sigm(acc[1][r]);
          float gv = tanh_f(acc[2][r]);
          float ov = sigm(acc[3][r]);
          float cn = fv * cc[n][r] + iv * gv;
          cc[n][r] = cn;
          float hn = ov * tanh_f(cn);
          hw[(r + off) * H + n * 16 + (lane & 15)] = (bf16_t)hn;
        }
      }
    }
    __syncthreads();
  }

  // h (bf16 in LDS) -> blockfeat fp32 [b][dir*128 + col], 8 elems per chunk
  for (int i = lane; i < 16 * H / 8; i += 32) {
    int e = 8 * i;
    int m = e / H, col = e % H;
    union { uint4 q; bf16_t b[8]; } pk;
    pk.q = ((const uint4*)hw)[i];
    float4 lo, hi;
    lo.x = (float)pk.b[0]; lo.y = (float)pk.b[1]; lo.z = (float)pk.b[2]; lo.w = (float)pk.b[3];
    hi.x = (float)pk.b[4]; hi.y = (float)pk.b[5]; hi.z = (float)pk.b[6]; hi.w = (float)pk.b[7];
    float* dstp = blockfeat + (size_t)(bbase + m) * (2 * H) + dir * H + col;
    *(float4*)(dstp)     = lo;
    *(float4*)(dstp + 4) = hi;
  }
}

// ---------------------------------------------------------------- final dot

__global__ void k_final(const float* __restrict__ blockfeat,
                        const float* __restrict__ global_params,
                        const float* __restrict__ final_W,
                        const float* __restrict__ final_b,
                        float* __restrict__ out) {
  __shared__ float red[64];
  int b = blockIdx.x, j = threadIdx.x;
  float p = 0.0f;
  for (int k = j; k < 2 * H; k += 64) p += blockfeat[(size_t)b * 2 * H + k] * final_W[k];
  if (j < P_GLOB) p += global_params[(size_t)b * P_GLOB + j] * final_W[2 * H + j];
  red[j] = p;
  __syncthreads();
  for (int s = 32; s > 0; s >>= 1) { if (j < s) red[j] += red[j + s]; __syncthreads(); }
  if (j == 0) out[b] = red[0] + final_b[0];
}

// ---------------------------------------------------------------- launcher

extern "C" void kernel_launch(void* const* d_in, const int* in_sizes, int n_in,
                              void* d_out, int out_size, void* d_ws, size_t ws_size,
                              hipStream_t stream) {
  const int*   tokens        = (const int*)d_in[0];
  const int*   token_lens    = (const int*)d_in[1];
  const int*   block_lens    = (const int*)d_in[2];
  const float* instr_params  = (const float*)d_in[3];
  const float* global_params = (const float*)d_in[4];
  const float* emb           = (const float*)d_in[5];
  const float* char_Wih_f    = (const float*)d_in[6];
  const float* char_Whh_f    = (const float*)d_in[7];
  const float* char_b_f      = (const float*)d_in[8];
  const float* char_Wih_b    = (const float*)d_in[9];
  const float* char_Whh_b    = (const float*)d_in[10];
  const float* char_b_b      = (const float*)d_in[11];
  const float* word_Wih_f    = (const float*)d_in[12];
  const float* word_Whh_f    = (const float*)d_in[13];
  const float* word_b_f      = (const float*)d_in[14];
  const float* word_Wih_b    = (const float*)d_in[15];
  const float* word_Whh_b    = (const float*)d_in[16];
  const float* word_b_b      = (const float*)d_in[17];
  const float* final_W       = (const float*)d_in[18];
  const float* final_b       = (const float*)d_in[19];
  float* out = (float*)d_out;

  // ---- workspace carve-up (bytes) ----
  char* ws = (char*)d_ws;
  size_t o = 0;
  auto take = [&](size_t bytes) { char* p = ws + o; o += (bytes + 255) & ~(size_t)255; return p; };
  bf16_t* cw_ih  = (bf16_t*)take(2 * (size_t)G4H * H * sizeof(bf16_t));      // [dir][512*128]
  bf16_t* cw_hh  = (bf16_t*)take(2 * (size_t)G4H * H * sizeof(bf16_t));
  bf16_t* ww_ih  = (bf16_t*)take(2 * (size_t)G4H * D_PAD * sizeof(bf16_t));  // [dir][512*320]
  bf16_t* ww_hh  = (bf16_t*)take(2 * (size_t)G4H * H * sizeof(bf16_t));
  bf16_t* emb_bf = (bf16_t*)take((size_t)N_TOKENS * H * sizeof(bf16_t));     // 1024 x 128
  bf16_t* full   = (bf16_t*)take((size_t)NI * D_PAD * sizeof(bf16_t));       // 16384 x 320
  float*  xW     = (float*)take(2 * (size_t)NI * G4H * sizeof(float));       // [dir][16384*512]
  int*    starts = (int*)take(NB * sizeof(int));
  int*    ends   = (int*)take(NB * sizeof(int));
  float*  bfeat  = (float*)take((size_t)NB * 2 * H * sizeof(float));         // 512 x 256
  (void)ws_size;

  // ---- conversions (fp32 -> bf16; word-Wih padded 296->320; emb table) ----
  {
    int n = G4H * H;                    // 65536
    dim3 g((n + 255) / 256), blk(256);
    k_cvt<<<g, blk, 0, stream>>>(char_Wih_f, cw_ih,            n);
    k_cvt<<<g, blk, 0, stream>>>(char_Wih_b, cw_ih + n,        n);
    k_cvt<<<g, blk, 0, stream>>>(char_Whh_f, cw_hh,            n);
    k_cvt<<<g, blk, 0, stream>>>(char_Whh_b, cw_hh + n,        n);
    k_cvt<<<g, blk, 0, stream>>>(word_Whh_f, ww_hh,            n);
    k_cvt<<<g, blk, 0, stream>>>(word_Whh_b, ww_hh + n,        n);
    int ne = N_TOKENS * H;              // 131072
    k_cvt<<<dim3((ne + 255) / 256), blk, 0, stream>>>(emb, emb_bf, ne);
    int np = G4H * D_PAD;               // 163840
    dim3 gp((np + 255) / 256);
    k_cvt_pad<<<gp, blk, 0, stream>>>(word_Wih_f, ww_ih,       G4H, D_WORD, D_PAD);
    k_cvt_pad<<<gp, blk, 0, stream>>>(word_Wih_b, ww_ih + np,  G4H, D_WORD, D_PAD);
  }

  // ---- block starts/ends + feature tail fill ----
  k_scan<<<1, NB, 0, stream>>>(block_lens, starts, ends);
  k_fill<<<NI, 64, 0, stream>>>(instr_params, global_params, ends, full);

  // ---- char-level BiLSTM (WMMA, weights LDS-resident, 8 waves/WG) ----
  {
    const size_t lds = (size_t)G4H * H * 2 * sizeof(bf16_t)   // Wih + Whh
                     + (size_t)G4H * sizeof(float)            // bias
                     + 8 * (size_t)16 * H * sizeof(bf16_t)    // h staging (8 waves)
                     + 8 * 16 * sizeof(int);                  // lens
    dim3 g(NI / 128, 2), blk(256);
    k_char_lstm<<<g, blk, lds, stream>>>(tokens, token_lens, emb_bf,
                                         cw_ih, cw_hh, char_b_f, char_b_b, full);
  }

  // ---- word-level input GEMM: xW = full @ Wih^T + b ----
  {
    dim3 g(NI / 16, 8, 2), blk(128);
    k_word_gemm<<<g, blk, 0, stream>>>(full, ww_ih, word_b_f, word_b_b, xW);
  }

  // ---- word-level BiLSTM recurrence (1 wave per WG, spread across WGPs) ----
  {
    const size_t lds = (size_t)G4H * H * sizeof(bf16_t)         // Whh
                     + (size_t)16 * H * sizeof(bf16_t)          // h staging
                     + 2 * 16 * sizeof(int);                    // starts + lens
    dim3 g(NB / 16, 2), blk(32);
    k_word_lstm<<<g, blk, lds, stream>>>(ww_hh, xW, starts, block_lens, bfeat);
  }

  // ---- final projection ----
  k_final<<<NB, 64, 0, stream>>>(bfeat, global_params, final_W, final_b, out);
}
